// SetGNNExtended_28269474742851
// MI455X (gfx1250) — compile-verified, matrix operand order checked by reading, not run on hardware
//
#include <hip/hip_runtime.h>
#include <hip/hip_bf16.h>

typedef __attribute__((ext_vector_type(16))) _Float16 v16h;
typedef __attribute__((ext_vector_type(8)))  float    v8f;

#define BN_SCALE 0.99999500003749973f  /* 1/sqrt(1+1e-5) */

__device__ __forceinline__ float wave_sum32(float v) {
  #pragma unroll
  for (int m = 16; m > 0; m >>= 1) v += __shfl_xor(v, m, 32);
  return v;
}

// ---------------------------------------------------------------------------
// Weight prep: fp32 row-major [K,Ncol] -> f16 col-major [Ncol,K]
// ---------------------------------------------------------------------------
__global__ void transpose_cvt_f16(const float* __restrict__ in, _Float16* __restrict__ out,
                                  int K, int Ncol) {
  int i = blockIdx.x * 256 + threadIdx.x;
  if (i >= K * Ncol) return;
  int k = i / Ncol, n = i - k * Ncol;
  out[(size_t)n * K + k] = (_Float16)in[i];
}

// ---------------------------------------------------------------------------
// Wave-level min reduction over dst_edges -> atomicMin
// ---------------------------------------------------------------------------
__global__ void idx_min_kernel(const int* __restrict__ d, int n, int* __restrict__ out) {
  int i = blockIdx.x * 256 + threadIdx.x;
  int v = (i < n) ? d[i] : 0x7FFFFFFF;
  #pragma unroll
  for (int m = 16; m > 0; m >>= 1) {
    int o = __shfl_xor(v, m, 32);
    v = (o < v) ? o : v;
  }
  if ((threadIdx.x & 31) == 0) atomicMin(out, v);
}

// ---------------------------------------------------------------------------
// LayerNorm over last dim W (128 or 256), optional relu, f16 output.
// One wave per row; lane owns float4 chunks.
// ---------------------------------------------------------------------------
template <int W, bool RELU>
__global__ void ln_kernel(const float* __restrict__ x, const float* __restrict__ g,
                          const float* __restrict__ b, _Float16* __restrict__ out, int R) {
  int r = blockIdx.x * 8 + threadIdx.y;
  if (r >= R) return;
  int lane = threadIdx.x;
  constexpr int C = W / 128;
  float4 v[C];
  float s = 0.f, sq = 0.f;
  const float* row = x + (size_t)r * W;
  #pragma unroll
  for (int c = 0; c < C; ++c) {
    v[c] = *(const float4*)(row + c * 128 + lane * 4);
    s  += v[c].x + v[c].y + v[c].z + v[c].w;
    sq += v[c].x * v[c].x + v[c].y * v[c].y + v[c].z * v[c].z + v[c].w * v[c].w;
  }
  s = wave_sum32(s);
  sq = wave_sum32(sq);
  const float mean = s * (1.0f / W);
  const float inv = rsqrtf(sq * (1.0f / W) - mean * mean + 1e-5f);
  _Float16* orow = out + (size_t)r * W;
  #pragma unroll
  for (int c = 0; c < C; ++c) {
    float e[4] = {v[c].x, v[c].y, v[c].z, v[c].w};
    #pragma unroll
    for (int j = 0; j < 4; ++j) {
      int col = c * 128 + lane * 4 + j;
      float y = (e[j] - mean) * inv * g[col] + b[col];
      if (RELU) y = fmaxf(y, 0.0f);
      orow[col] = (_Float16)y;
    }
  }
}

// ---------------------------------------------------------------------------
// WMMA GEMM: C[rows,ncols] = (relu?)(A_f16[rows,K] x Bt_f16[ncols,K]^T + bias)
// wave = one 16x16 tile, 8 waves/block -> 16 rows x 128 cols per block.
// blockIdx.x = row tile, blockIdx.y = group of 128 cols. K % 32 == 0.
// ---------------------------------------------------------------------------
template <bool RELU>
__global__ void gemm_wmma_f16(const _Float16* __restrict__ A, const _Float16* __restrict__ Bt,
                              const float* __restrict__ bias, float* __restrict__ C,
                              int K, int ncols) {
  const int lane = threadIdx.x;
  const int wave = threadIdx.y;
  const int l16 = lane & 15;
  const int half = lane >> 4;
  const int rowbase = blockIdx.x * 16;
  const int colbase = (blockIdx.y * 8 + wave) * 16;

  const _Float16* arow = A + (size_t)(rowbase + l16) * K + half * 16;
  const _Float16* bcol = Bt + (size_t)(colbase + l16) * K + half * 16;

  v8f acc = {};
  for (int kc = 0; kc < K; kc += 32) {
    if (kc + 32 < K) __builtin_prefetch((const void*)(arow + kc + 32), 0, 1);
    v16h a = *(const v16h*)(arow + kc);
    v16h b = *(const v16h*)(bcol + kc);
    acc = __builtin_amdgcn_wmma_f32_16x16x32_f16(false, a, false, b, (short)0, acc,
                                                 false, false);
  }

  const int col = colbase + l16;
  const float bv = bias[col];
  float* crow = C + (size_t)(rowbase + (half ? 8 : 0)) * ncols + col;
  #pragma unroll
  for (int vr = 0; vr < 8; ++vr) {
    float r = acc[vr] + bv;
    if (RELU) r = fmaxf(r, 0.0f);
    crow[(size_t)vr * ncols] = r;
  }
}

// ---------------------------------------------------------------------------
// msg = norm[e] * h[gidx[e]] ; atomic scatter-add into agg[sidx[e]] + counts.
// One wave per edge, float4 per lane.
// ---------------------------------------------------------------------------
__global__ void gather_scatter(const float* __restrict__ h, const float* __restrict__ norm,
                               const int* __restrict__ gidx, int gsub,
                               const int* __restrict__ sidx, int ssub,
                               const int* __restrict__ dminp,
                               float* __restrict__ agg, float* __restrict__ cnt, int E) {
  int e = blockIdx.x * 8 + threadIdx.y;
  if (e >= E) return;
  const int dmin = *dminp;
  const int gi = gidx[e] - (gsub ? dmin : 0);
  const int si = sidx[e] - (ssub ? dmin : 0);
  const float w = norm[e];
  const int lane = threadIdx.x;
  const float4 v = *(const float4*)(h + (size_t)gi * 128 + lane * 4);
  float* dp = agg + (size_t)si * 128 + lane * 4;
  atomicAdd(dp + 0, w * v.x);
  atomicAdd(dp + 1, w * v.y);
  atomicAdd(dp + 2, w * v.z);
  atomicAdd(dp + 3, w * v.w);
  if (lane == 0) atomicAdd(&cnt[si], 1.0f);
}

// ---------------------------------------------------------------------------
// agg / max(cnt,1) -> f16 (dec GEMM A operand)
// ---------------------------------------------------------------------------
__global__ void seg_mean_cvt(const float* __restrict__ agg, const float* __restrict__ cnt,
                             int R, _Float16* __restrict__ out) {
  int i = blockIdx.x * 256 + threadIdx.x;
  if (i >= R * 128) return;
  const float c = fmaxf(cnt[i >> 7], 1.0f);
  out[i] = (_Float16)(agg[i] / c);
}

// ---------------------------------------------------------------------------
// center_scale stage 1: column sums + total sum-of-squares (block-local
// accumulation over grid-stride rows, then a few atomics per block).
// stats[0..127] = colsum, stats[128] = sumsq.
// ---------------------------------------------------------------------------
__global__ void reduce_stats(const float* __restrict__ x, int R, float* __restrict__ stats) {
  const int lane = threadIdx.x;
  float c0 = 0.f, c1 = 0.f, c2 = 0.f, c3 = 0.f, sq = 0.f;
  for (int r = blockIdx.x * 8 + threadIdx.y; r < R; r += gridDim.x * 8) {
    const float4 v = *(const float4*)(x + (size_t)r * 128 + lane * 4);
    c0 += v.x; c1 += v.y; c2 += v.z; c3 += v.w;
    sq += v.x * v.x + v.y * v.y + v.z * v.z + v.w * v.w;
  }
  atomicAdd(&stats[lane * 4 + 0], c0);
  atomicAdd(&stats[lane * 4 + 1], c1);
  atomicAdd(&stats[lane * 4 + 2], c2);
  atomicAdd(&stats[lane * 4 + 3], c3);
  sq = wave_sum32(sq);
  if (lane == 0) atomicAdd(&stats[128], sq);
}

// stage 2: stats[0..127] -> col means, stats[129] = 1/sqrt(eps + sumsq/R - sum(mean^2))
__global__ void finalize_stats(float* __restrict__ stats, float invR) {
  __shared__ float sm[128];
  const int t = threadIdx.x;
  const float mean = stats[t] * invR;
  stats[t] = mean;
  sm[t] = mean * mean;
  __syncthreads();
  #pragma unroll
  for (int s = 64; s > 0; s >>= 1) {
    if (t < s) sm[t] += sm[t + s];
    __syncthreads();
  }
  if (t == 0) {
    const float sumsq = stats[128];
    stats[129] = rsqrtf(1e-5f + sumsq * invR - sm[0]);
  }
}

// stage 3: fan out centered value to (optional) fp32 output, next-x, f16 hcat slice
__global__ void apply_center(const float* __restrict__ in, const float* __restrict__ stats,
                             int R, float* __restrict__ center_out, float* __restrict__ xnext,
                             _Float16* __restrict__ f16dst, int f16ld, int f16off) {
  int i = blockIdx.x * 256 + threadIdx.x;
  if (i >= R * 128) return;
  const int c = i & 127;
  const int r = i >> 7;
  const float v = (in[i] - stats[c]) * stats[129];
  if (center_out) center_out[i] = v;
  if (xnext) xnext[i] = fmaxf(v * BN_SCALE, 0.0f);
  if (f16dst) f16dst[(size_t)r * f16ld + f16off + c] = (_Float16)v;
}

// ---------------------------------------------------------------------------
// Final 256x2 head: wave per row, shfl reduction.
// ---------------------------------------------------------------------------
__global__ void final_head(const _Float16* __restrict__ h, const float* __restrict__ W2,
                           const float* __restrict__ b2, float* __restrict__ out, int R) {
  int r = blockIdx.x * 8 + threadIdx.y;
  if (r >= R) return;
  const int lane = threadIdx.x;
  float a0 = 0.f, a1 = 0.f;
  const _Float16* row = h + (size_t)r * 256;
  #pragma unroll
  for (int k = lane; k < 256; k += 32) {
    const float v = (float)row[k];
    a0 += v * W2[k * 2 + 0];
    a1 += v * W2[k * 2 + 1];
  }
  a0 = wave_sum32(a0);
  a1 = wave_sum32(a1);
  if (lane == 0) {
    out[(size_t)r * 2 + 0] = a0 + b2[0];
    out[(size_t)r * 2 + 1] = a1 + b2[1];
  }
}

// ---------------------------------------------------------------------------
extern "C" void kernel_launch(void* const* d_in, const int* in_sizes, int n_in,
                              void* d_out, int out_size, void* d_ws, size_t ws_size,
                              hipStream_t stream) {
  constexpr int N = 100000, M = 20000, E = 1000000, H = 128;

  const float* x_in     = (const float*)d_in[0];
  const int*   src      = (const int*)d_in[1];
  const int*   dste     = (const int*)d_in[2];
  const float* norm     = (const float*)d_in[3];
  const float* enc_ln_g = (const float*)d_in[4];
  const float* enc_ln_b = (const float*)d_in[5];
  const float* enc_W    = (const float*)d_in[6];
  const float* enc_b    = (const float*)d_in[7];
  const float* dec_W    = (const float*)d_in[8];
  const float* dec_b    = (const float*)d_in[9];
  const float* cls_W1   = (const float*)d_in[10];
  const float* cls_b1   = (const float*)d_in[11];
  const float* cls_ln_g = (const float*)d_in[12];
  const float* cls_ln_b = (const float*)d_in[13];
  const float* cls_W2   = (const float*)d_in[14];
  const float* cls_b2   = (const float*)d_in[15];

  // workspace carve (bump allocator, 256B aligned)
  char* p = (char*)d_ws;
  auto carve = [&](size_t bytes) -> char* {
    char* r = p;
    p += (bytes + 255) & ~(size_t)255;
    return r;
  };
  _Float16* F16A  = (_Float16*)carve((size_t)N * 128 * 2);  // LN / mean f16 operand
  float*    BUF1  = (float*)carve((size_t)N * 128 * 4);     // enc-out h / dec-out
  float*    BUF2  = (float*)carve((size_t)N * 128 * 4);     // agg accumulator
  float*    X32   = (float*)carve((size_t)N * 128 * 4);     // running x
  float*    CNT   = (float*)carve((size_t)N * 4);
  _Float16* HCAT  = (_Float16*)carve((size_t)M * 384 * 2);
  _Float16* ENCWT = (_Float16*)carve((size_t)5 * 128 * 128 * 2);
  _Float16* DECWT = (_Float16*)carve((size_t)5 * 128 * 128 * 2);
  _Float16* W1T   = (_Float16*)carve((size_t)384 * 256 * 2);
  float*    STATS = (float*)carve(1024);
  int*      DMIN  = (int*)carve(256);

  float* out_score = (float*)d_out;                    // [M,2]
  float* out_efeat = out_score + (size_t)M * 2;        // [M,128]
  float* out_nfeat = out_efeat + (size_t)M * 128;      // [N,128]

  const dim3 blk(32, 8);

  // ---- prep: dst min + weight transposes to f16 col-major
  hipMemsetAsync(DMIN, 0x7F, 4, stream);
  idx_min_kernel<<<(E + 255) / 256, 256, 0, stream>>>(dste, E, DMIN);
  for (int k = 0; k < 5; ++k) {
    transpose_cvt_f16<<<(16384 + 255) / 256, 256, 0, stream>>>(enc_W + (size_t)k * 16384,
                                                               ENCWT + (size_t)k * 16384, 128, 128);
    transpose_cvt_f16<<<(16384 + 255) / 256, 256, 0, stream>>>(dec_W + (size_t)k * 16384,
                                                               DECWT + (size_t)k * 16384, 128, 128);
  }
  transpose_cvt_f16<<<(384 * 256 + 255) / 256, 256, 0, stream>>>(cls_W1, W1T, 384, 256);

  // ---- one half-conv + center_scale
  auto conv = [&](int k, const float* feat, int RS, int RD,
                  const int* gidx, int gsub, const int* sidx, int ssub,
                  float* center_out, float* xnext, int hcat_off) {
    ln_kernel<128, false><<<(RS + 7) / 8, blk, 0, stream>>>(feat, enc_ln_g + k * H,
                                                            enc_ln_b + k * H, F16A, RS);
    gemm_wmma_f16<true><<<dim3(RS / 16, 1), blk, 0, stream>>>(F16A, ENCWT + (size_t)k * 16384,
                                                              enc_b + k * H, BUF1, 128, 128);
    hipMemsetAsync(BUF2, 0, (size_t)RD * 128 * 4, stream);
    hipMemsetAsync(CNT, 0, (size_t)RD * 4, stream);
    gather_scatter<<<(E + 7) / 8, blk, 0, stream>>>(BUF1, norm, gidx, gsub, sidx, ssub,
                                                    DMIN, BUF2, CNT, E);
    seg_mean_cvt<<<((size_t)RD * 128 + 255) / 256, 256, 0, stream>>>(BUF2, CNT, RD, F16A);
    gemm_wmma_f16<true><<<dim3(RD / 16, 1), blk, 0, stream>>>(F16A, DECWT + (size_t)k * 16384,
                                                              dec_b + k * H, BUF1, 128, 128);
    hipMemsetAsync(STATS, 0, 1024, stream);
    reduce_stats<<<256, blk, 0, stream>>>(BUF1, RD, STATS);
    finalize_stats<<<1, 128, 0, stream>>>(STATS, 1.0f / (float)RD);
    apply_center<<<((size_t)RD * 128 + 255) / 256, 256, 0, stream>>>(
        BUF1, STATS, RD, center_out, xnext,
        (hcat_off >= 0) ? HCAT : (_Float16*)nullptr, 384, (hcat_off >= 0) ? hcat_off : 0);
  };

  // layer schedule (mirrors reference unroll)
  conv(0, x_in, N, M, src, 0, dste, 1, nullptr,   X32,     0);   // V2E0 -> vec[0], x
  conv(1, X32,  M, N, dste, 1, src, 0, nullptr,   X32,    -1);   // E2V0 -> x
  conv(2, X32,  N, M, src, 0, dste, 1, nullptr,   X32,   128);   // V2E1 -> vec[1], x
  conv(3, X32,  M, N, dste, 1, src, 0, out_nfeat, X32,    -1);   // E2V1 -> node_feat, x
  conv(4, X32,  N, M, src, 0, dste, 1, out_efeat, nullptr, 256); // V2E2 -> edge_feat/vec[2]

  // ---- classifier: [M,384] x [384,256] + b1 -> LN(256)+relu -> [256,2] head
  float*    CLSH32 = BUF1;             // reuse
  _Float16* CLSH16 = (_Float16*)BUF2;  // reuse
  gemm_wmma_f16<false><<<dim3(M / 16, 2), blk, 0, stream>>>(HCAT, W1T, cls_b1, CLSH32, 384, 256);
  ln_kernel<256, true><<<(M + 7) / 8, blk, 0, stream>>>(CLSH32, cls_ln_g, cls_ln_b, CLSH16, M);
  final_head<<<(M + 7) / 8, blk, 0, stream>>>(CLSH16, cls_W2, cls_b2, out_score, M);
}
